// TemporalFusionTransformer_85495618994601
// MI455X (gfx1250) — matrix operand order repaired and checked by reading, not compiled, not run on hardware
//
#include <hip/hip_runtime.h>
#include <hip/hip_bf16.h>
#include <math.h>

#define B_    512
#define F_    9
#define H_    128
#define NH_   4
#define OUT_  6
#define TENC_ 168
#define TDEC_ 24
#define FH_   (F_*H_)        // 1152
#define MENC_ (B_*TENC_)     // 86016
#define MDEC_ (B_*TDEC_)     // 12288

typedef __attribute__((ext_vector_type(16))) _Float16     v16h;
typedef __attribute__((ext_vector_type(8)))  float        v8f;
typedef __attribute__((ext_vector_type(4)))  unsigned int v4u;

__device__ __forceinline__ float sigm(float x){ return 1.f/(1.f+__expf(-x)); }
__device__ __forceinline__ float eluf(float x){ return x>0.f ? x : __expf(x)-1.f; }

// ---- WMMA fragment helpers (CDNA5 16x16x32 f16, wave32) ----
// A: 16x32 (MxK) tile, LDS row-major (lda halves). Lane layout (ISA 7.12.2):
//   m = lane%16, grp = lane/16; halves 0..7 <- k = grp*8..grp*8+7,
//   halves 8..15 <- k = 16+grp*8.. => two 16B LDS loads.
__device__ __forceinline__ v16h load_a(const _Float16* a, int lda) {
  const int lane = threadIdx.x & 31;
  const int m = lane & 15, grp = lane >> 4;
  const _Float16* p = a + m*lda + grp*8;
  union { v16h h; v4u u[2]; } r;
  r.u[0] = *(const v4u*)(p);
  r.u[1] = *(const v4u*)(p + 16);
  return r.h;
}
// B: 32x16 (KxN) tile stored N-MAJOR in LDS: b[n*ldk + k]. Lane layout:
//   n = lane%16, grp = lane/16; halves 0..7 <- k = grp*8..grp*8+7, 8..15 <- k = 16+grp*8..
//   => two 16B LDS loads.
__device__ __forceinline__ v16h load_bn(const _Float16* b, int ldk) {
  const int lane = threadIdx.x & 31;
  const int n = lane & 15, grp = lane >> 4;
  const _Float16* p = b + n*ldk + grp*8;
  union { v16h h; v4u u[2]; } r;
  r.u[0] = *(const v4u*)(p);
  r.u[1] = *(const v4u*)(p + 16);
  return r.h;
}
__device__ __forceinline__ v8f wmma_f16(v16h a, v16h b, v8f c) {
  return __builtin_amdgcn_wmma_f32_16x16x32_f16(false, a, false, b, (short)0, c, false, false);
}
// C/D: lane holds D[m = r + 8*(lane/16)][n = lane%16] in acc[r]

__device__ __forceinline__ void store_pk(_Float16* dst, float v0, float v1) {
  union { _Float16 h[2]; unsigned u; } pk;
  pk.h[0] = (_Float16)v0; pk.h[1] = (_Float16)v1;
  *(unsigned*)dst = pk.u;
}

// ======== Generic GEMM: out = act(A_f16[MxK] @ B + bias0 + bias1) ========
// btrans=0: B(k,n)=Bw[k*ldb+n]; btrans=1: B(k,n)=Bw[n*ldb+k]
// grid: (M/16, N/128), block 256 (8 waves; wave w owns 16-col N-subtile)
template<int ACT>
__global__ __launch_bounds__(256) void gemm16(
    const _Float16* __restrict__ A, int M, int K,
    const float* __restrict__ Bw, int ldb, int btrans,
    const float* __restrict__ bias0, const float* __restrict__ bias1,
    float* __restrict__ outF, _Float16* __restrict__ outH, int ldo)
{
  __shared__ __align__(16) _Float16 As[16*32];        // row-major [m][k]
  __shared__ __align__(16) _Float16 Bs[8][16*32];     // N-major  [n][k]
  const int tid = threadIdx.x, wave = tid>>5, lane = tid&31;
  const int m0 = blockIdx.x*16;
  const int n0 = blockIdx.y*128 + wave*16;
  const int ar = tid>>4, ac2 = (tid&15)*2;            // A staging map
  v8f acc = {};
  for (int kk = 0; kk < K; kk += 32) {
    __syncthreads();
    // stage A tile (one packed 32-bit copy per thread)
    *(unsigned*)(As + ar*32 + ac2) =
        *(const unsigned*)(A + (size_t)(m0 + ar)*K + kk + ac2);
    if (kk + 32 < K)
      __builtin_prefetch(A + (size_t)(m0 + ar)*K + kk + 32 + ac2, 0, 0);
    // stage B tile N-major (wave local), f32->f16
    for (int i = lane; i < 256; i += 32) {
      int n = i>>4, k2 = (i&15)*2;
      float v0 = btrans ? Bw[(size_t)(n0+n)*ldb + kk+k2]
                        : Bw[(size_t)(kk+k2)*ldb + n0+n];
      float v1 = btrans ? Bw[(size_t)(n0+n)*ldb + kk+k2+1]
                        : Bw[(size_t)(kk+k2+1)*ldb + n0+n];
      store_pk(&Bs[wave][n*32 + k2], v0, v1);
    }
    __syncthreads();
    acc = wmma_f16(load_a(As, 32), load_bn(Bs[wave], 32), acc);
  }
  const int n = lane & 15, grp = lane >> 4;
#pragma unroll
  for (int r = 0; r < 8; ++r) {
    int m = m0 + r + 8*grp;
    int c = n0 + n;
    float v = acc[r];
    if (bias0) v += bias0[c];
    if (bias1) v += bias1[c];
    if (ACT == 1) v = eluf(v);
    if (ACT == 2) v = sigm(v);
    if (outF) outF[(size_t)m*ldo + c] = v;
    if (outH) outH[(size_t)m*ldo + c] = (_Float16)v;
  }
}

// ======== VSN: per-feature GRN fused (h1 gen + dual WMMA GEMM + LN) ========
// grid: (MENC/16, F), block 256
__global__ __launch_bounds__(256) void vsn_kernel(
    const float* __restrict__ x,
    const float* __restrict__ w1,  const float* __restrict__ b1,
    const float* __restrict__ fc2w,const float* __restrict__ fc2b,
    const float* __restrict__ gw,  const float* __restrict__ gb,
    const float* __restrict__ sw,  const float* __restrict__ sb,
    const float* __restrict__ lns, const float* __restrict__ lnb,
    _Float16* __restrict__ flat16)
{
  __shared__ float    xs[16];
  __shared__ __align__(16) _Float16 H1[16*128];       // row-major [m][k]
  __shared__ __align__(16) _Float16 Bs[8][2][16*32];  // N-major  [n][k]
  __shared__ float    VAL[16*128];
  const int tid = threadIdx.x, wave = tid>>5, lane = tid&31;
  const int m0 = blockIdx.x*16, f = blockIdx.y;
  if (tid < 16) {
    int mg = m0 + tid;
    int b = mg / TENC_, t = mg % TENC_;
    xs[tid] = x[((size_t)b*(TENC_+TDEC_) + t)*F_ + f];
  }
  __syncthreads();
  for (int i = tid; i < 16*128; i += 256) {
    int m = i>>7, h = i&127;
    H1[i] = (_Float16)eluf(xs[m]*w1[f*H_ + h] + b1[f*H_ + h]);
  }
  __syncthreads();
  const int n0 = wave*16;
  const float* W2 = fc2w + (size_t)f*H_*H_;
  const float* WG = gw   + (size_t)f*H_*H_;
  v8f ah = {}, ag = {};
  for (int kk = 0; kk < 128; kk += 32) {
    for (int i = lane; i < 256; i += 32) {            // stage both B tiles
      int n = i>>4, k2 = (i&15)*2;
      store_pk(&Bs[wave][0][n*32 + k2],
               W2[(size_t)(kk+k2)*H_ + n0+n], W2[(size_t)(kk+k2+1)*H_ + n0+n]);
      store_pk(&Bs[wave][1][n*32 + k2],
               WG[(size_t)(kk+k2)*H_ + n0+n], WG[(size_t)(kk+k2+1)*H_ + n0+n]);
    }
    v16h a = load_a(H1 + kk, 128);
    ah = wmma_f16(a, load_bn(Bs[wave][0], 32), ah);
    ag = wmma_f16(a, load_bn(Bs[wave][1], 32), ag);
  }
  const int n = lane & 15, grp = lane >> 4;
#pragma unroll
  for (int r = 0; r < 8; ++r) {
    int m = r + 8*grp, c = n0 + n;
    float hv = ah[r] + fc2b[f*H_ + c];
    float gv = sigm(ag[r] + gb[f*H_ + c]);
    float sk = xs[m]*sw[f*H_ + c] + sb[f*H_ + c];
    VAL[m*128 + c] = sk + hv*gv;
  }
  __syncthreads();
  for (int rr = 0; rr < 2; ++rr) {                    // wave handles rows 2w, 2w+1
    int m = wave*2 + rr;
    float v[4], s = 0.f, sq = 0.f;
#pragma unroll
    for (int q = 0; q < 4; ++q) {
      v[q] = VAL[m*128 + lane + 32*q];
      s += v[q]; sq += v[q]*v[q];
    }
#pragma unroll
    for (int i = 16; i > 0; i >>= 1) { s += __shfl_xor(s,i); sq += __shfl_xor(sq,i); }
    float mean = s*(1.f/128.f), var = sq*(1.f/128.f) - mean*mean;
    float rstd = rsqrtf(var + 1e-5f);
    int mg = m0 + m;
#pragma unroll
    for (int q = 0; q < 4; ++q) {
      int c = lane + 32*q;
      flat16[(size_t)mg*FH_ + f*H_ + c] =
        (_Float16)((v[q]-mean)*rstd*lns[f*H_+c] + lnb[f*H_+c]);
    }
  }
}

// ======== LN combine: out = LN(skip + h*g) over H=128 ========
__global__ __launch_bounds__(256) void ln_combine(
    const float* __restrict__ skip, const float* __restrict__ hb,
    const float* __restrict__ gbuf, const float* __restrict__ s,
    const float* __restrict__ bv, int M,
    _Float16* __restrict__ outH, float* __restrict__ outF)
{
  const int wave = threadIdx.x>>5, lane = threadIdx.x&31;
  const int row = blockIdx.x*8 + wave;
  if (row >= M) return;
  size_t base = (size_t)row*128;
  float v[4], sm = 0.f, sq = 0.f;
#pragma unroll
  for (int q = 0; q < 4; ++q) {
    int c = lane + 32*q;
    v[q] = skip[base+c] + hb[base+c]*gbuf[base+c];
    sm += v[q]; sq += v[q]*v[q];
  }
#pragma unroll
  for (int i = 16; i > 0; i >>= 1) { sm += __shfl_xor(sm,i); sq += __shfl_xor(sq,i); }
  float mean = sm*(1.f/128.f), var = sq*(1.f/128.f) - mean*mean;
  float rstd = rsqrtf(var + 1e-5f);
#pragma unroll
  for (int q = 0; q < 4; ++q) {
    int c = lane + 32*q;
    float o = (v[q]-mean)*rstd*s[c] + bv[c];
    if (outH) outH[base+c] = (_Float16)o;
    if (outF) outF[base+c] = o;
  }
}

// ======== Persistent LSTM recurrence ========
// 16 batch rows per WG; Whh resident in LDS (f16, N-major, 128KB); h/c in LDS.
// Wave w owns column slice [w*16, w*16+16) of ALL FOUR gates => pure register
// epilogue, no cross-wave gate exchange. G: precomputed x@Wih.T + bih + bhh.
__global__ __launch_bounds__(256) void lstm_rec(
    const float* __restrict__ G, const float* __restrict__ Whh,
    const float* __restrict__ h0, const float* __restrict__ c0, int T,
    _Float16* __restrict__ seq, float* __restrict__ hT, float* __restrict__ cT)
{
  __shared__ __align__(16) _Float16 WhhS[512*128];    // [n][k] f16, 128KB
  __shared__ __align__(16) _Float16 Hs[16*128];
  __shared__ float Cs[16*128];
  const int tid = threadIdx.x, wave = tid>>5, lane = tid&31;
  const int b0 = blockIdx.x*16;
  // resident recurrent weights: Whh (512x128 f32, row n = out, col k = in)
  for (int i = tid; i < 512*128/4; i += 256) {
    const float4 w = ((const float4*)Whh)[i];
    _Float16* d = WhhS + (size_t)i*4;
    d[0] = (_Float16)w.x; d[1] = (_Float16)w.y;
    d[2] = (_Float16)w.z; d[3] = (_Float16)w.w;
  }
  for (int i = tid; i < 16*128; i += 256) {
    int m = i>>7, j = i&127;
    Hs[i] = (_Float16)(h0 ? h0[(size_t)(b0+m)*128 + j] : 0.f);
    Cs[i] = c0 ? c0[(size_t)(b0+m)*128 + j] : 0.f;
  }
  __syncthreads();
  const int n = lane&15, grp = lane>>4, c = wave*16 + (lane&15);
  for (int t = 0; t < T; ++t) {
    v8f ai = {}, af = {}, ag = {}, ao = {};
#pragma unroll
    for (int kk = 0; kk < 128; kk += 32) {
      v16h a = load_a(Hs + kk, 128);
      ai = wmma_f16(a, load_bn(WhhS + (size_t)(  0 + wave*16)*128 + kk, 128), ai);
      af = wmma_f16(a, load_bn(WhhS + (size_t)(128 + wave*16)*128 + kk, 128), af);
      ag = wmma_f16(a, load_bn(WhhS + (size_t)(256 + wave*16)*128 + kk, 128), ag);
      ao = wmma_f16(a, load_bn(WhhS + (size_t)(384 + wave*16)*128 + kk, 128), ao);
    }
    if (t + 1 < T)                                     // prefetch next step gates
      __builtin_prefetch(G + ((size_t)(b0 + (tid>>4))*T + t + 1)*512 + n*32, 0, 0);
    __syncthreads();                                   // all Hs reads complete
#pragma unroll
    for (int r = 0; r < 8; ++r) {
      int m = r + 8*grp;
      size_t grow = ((size_t)(b0+m)*T + t)*512;
      float gi = ai[r] + G[grow +       c];
      float gf = af[r] + G[grow + 128 + c];
      float gg = ag[r] + G[grow + 256 + c];
      float go = ao[r] + G[grow + 384 + c];
      float cv = sigm(gf)*Cs[m*128+c] + sigm(gi)*tanhf(gg);
      float hv = sigm(go)*tanhf(cv);
      Cs[m*128+c] = cv;
      Hs[m*128+c] = (_Float16)hv;
      seq[((size_t)(b0+m)*T + t)*128 + c] = (_Float16)hv;
    }
    __syncthreads();
  }
  for (int i = tid; i < 16*128; i += 256) {
    int m = i>>7, j = i&127;
    hT[(size_t)(b0+m)*128 + j] = (float)Hs[i];
    cT[(size_t)(b0+m)*128 + j] = Cs[i];
  }
}

// ======== dec_seq init: repeat enc_out[:, -1, :] ========
__global__ void repeat_last(const _Float16* __restrict__ enc, _Float16* __restrict__ dst) {
  int idx = blockIdx.x*256 + threadIdx.x;
  if (idx >= MDEC_*128) return;
  int m = idx >> 7, j = idx & 127;
  int b = m / TDEC_;
  dst[idx] = enc[((size_t)b*TENC_ + (TENC_-1))*128 + j];
}

// ======== attention core: scores + softmax + context (head per wave) ========
__global__ __launch_bounds__(128) void attn_core(
    const _Float16* __restrict__ q, const _Float16* __restrict__ k,
    const _Float16* __restrict__ v, _Float16* __restrict__ ctx)
{
  __shared__ float sc[NH_][TENC_];
  const int head = threadIdx.x>>5, lane = threadIdx.x&31;
  const int row = blockIdx.x;               // b*TDEC + qi
  const int b = row / TDEC_;
  const float qd = (float)q[(size_t)row*128 + head*32 + lane];
  for (int kk = 0; kk < TENC_; ++kk) {
    float p = qd * (float)k[((size_t)b*TENC_ + kk)*128 + head*32 + lane];
#pragma unroll
    for (int i = 16; i > 0; i >>= 1) p += __shfl_xor(p, i);
    if (lane == 0) sc[head][kk] = p * 0.17677669529663689f;  // 1/sqrt(32)
  }
  float mx = -1e30f;
  for (int kk = lane; kk < TENC_; kk += 32) mx = fmaxf(mx, sc[head][kk]);
#pragma unroll
  for (int i = 16; i > 0; i >>= 1) mx = fmaxf(mx, __shfl_xor(mx, i));
  float sm = 0.f;
  for (int kk = lane; kk < TENC_; kk += 32) {
    float e = __expf(sc[head][kk] - mx);
    sc[head][kk] = e; sm += e;
  }
#pragma unroll
  for (int i = 16; i > 0; i >>= 1) sm += __shfl_xor(sm, i);
  float inv = 1.f/sm, acc = 0.f;
  for (int kk = 0; kk < TENC_; ++kk)
    acc += sc[head][kk] * (float)v[((size_t)b*TENC_ + kk)*128 + head*32 + lane];
  ctx[(size_t)row*128 + head*32 + lane] = (_Float16)(acc*inv);
}

// ======== final: out = post @ fc_out_w + b (N=6, VALU) ========
__global__ void final_out(const _Float16* __restrict__ post,
                          const float* __restrict__ w, const float* __restrict__ bo,
                          float* __restrict__ out)
{
  int idx = blockIdx.x*256 + threadIdx.x;
  if (idx >= MDEC_*OUT_) return;
  int m = idx / OUT_, o = idx % OUT_;
  float acc = bo[o];
  for (int j = 0; j < 128; ++j)
    acc += (float)post[(size_t)m*128 + j] * w[j*OUT_ + o];
  out[idx] = acc;
}

extern "C" void kernel_launch(void* const* d_in, const int* in_sizes, int n_in,
                              void* d_out, int out_size, void* d_ws, size_t ws_size,
                              hipStream_t stream) {
  const float* x          = (const float*)d_in[0];
  const float* vsn_fc1_w  = (const float*)d_in[1];
  const float* vsn_fc1_b  = (const float*)d_in[2];
  const float* vsn_fc2_w  = (const float*)d_in[3];
  const float* vsn_fc2_b  = (const float*)d_in[4];
  const float* vsn_gate_w = (const float*)d_in[5];
  const float* vsn_gate_b = (const float*)d_in[6];
  const float* vsn_skip_w = (const float*)d_in[7];
  const float* vsn_skip_b = (const float*)d_in[8];
  const float* vsn_ln_s   = (const float*)d_in[9];
  const float* vsn_ln_b   = (const float*)d_in[10];
  const float* comb_fc1_w = (const float*)d_in[11];
  const float* comb_fc1_b = (const float*)d_in[12];
  const float* comb_fc2_w = (const float*)d_in[13];
  const float* comb_fc2_b = (const float*)d_in[14];
  const float* comb_gate_w= (const float*)d_in[15];
  const float* comb_gate_b= (const float*)d_in[16];
  const float* comb_skip_w= (const float*)d_in[17];
  const float* comb_skip_b= (const float*)d_in[18];
  const float* comb_ln_s  = (const float*)d_in[19];
  const float* comb_ln_b  = (const float*)d_in[20];
  const float* enc_Wih    = (const float*)d_in[21];
  const float* enc_Whh    = (const float*)d_in[22];
  const float* enc_bih    = (const float*)d_in[23];
  const float* enc_bhh    = (const float*)d_in[24];
  const float* dec_Wih    = (const float*)d_in[25];
  const float* dec_Whh    = (const float*)d_in[26];
  const float* dec_bih    = (const float*)d_in[27];
  const float* dec_bhh    = (const float*)d_in[28];
  const float* qkv_w      = (const float*)d_in[29];
  const float* qkv_b      = (const float*)d_in[30];
  const float* attn_out_w = (const float*)d_in[31];
  const float* attn_out_b = (const float*)d_in[32];
  const float* post_fc1_w = (const float*)d_in[33];
  const float* post_fc1_b = (const float*)d_in[34];
  const float* post_fc2_w = (const float*)d_in[35];
  const float* post_fc2_b = (const float*)d_in[36];
  const float* post_gate_w= (const float*)d_in[37];
  const float* post_gate_b= (const float*)d_in[38];
  const float* post_ln_s  = (const float*)d_in[39];
  const float* post_ln_b  = (const float*)d_in[40];
  const float* fc_out_w   = (const float*)d_in[41];
  const float* fc_out_b   = (const float*)d_in[42];
  (void)in_sizes; (void)n_in; (void)out_size; (void)ws_size;

  // ---- workspace plan (phase-based reuse) ----
  char* ws = (char*)d_ws;
  size_t off = 0;
  auto region = [&](size_t bytes){ size_t r = off; off += (bytes + 255) & ~(size_t)255; return r; };
  size_t BIG0 = region((size_t)MENC_*FH_*2);   // flat16 -> Gbuf -> v16
  size_t RB1  = region((size_t)MENC_*128*4);   // hF -> enc_out16
  size_t RB2  = region((size_t)MENC_*128*4);   // gF -> k16
  size_t RB3  = region((size_t)MENC_*128*4);   // sF -> attn pool
  size_t RM0  = region((size_t)MENC_*128*2);   // h1c16 -> seq0 -> dec bufs
  size_t RM1  = region((size_t)MENC_*128*2);   // comb16
  size_t RS0  = region((size_t)6*B_*128*4);    // hT/cT states

  _Float16* flat16 = (_Float16*)(ws + BIG0);
  float*    Gbuf   = (float*)   (ws + BIG0);
  _Float16* v16    = (_Float16*)(ws + BIG0);
  float*    hF     = (float*)   (ws + RB1);
  _Float16* enc1   = (_Float16*)(ws + RB1);    // enc_out f16
  float*    gF     = (float*)   (ws + RB2);
  _Float16* k16    = (_Float16*)(ws + RB2);
  float*    sF     = (float*)   (ws + RB3);
  _Float16* q16    = (_Float16*)(ws + RB3);
  _Float16* ctx16  = (_Float16*)(ws + RB3 +  3145728);
  float*    attnF  = (float*)   (ws + RB3 +  6291456);
  _Float16* attn16 = (_Float16*)(ws + RB3 + 12582912);
  _Float16* ph116  = (_Float16*)(ws + RB3 + 15728640);
  float*    phF    = (float*)   (ws + RB3 + 18874368);
  float*    pgF    = (float*)   (ws + RB3 + 25165824);
  _Float16* post16 = (_Float16*)(ws + RB3 + 31457280);
  _Float16* h1c16  = (_Float16*)(ws + RM0);
  _Float16* seq0   = (_Float16*)(ws + RM0);
  _Float16* decin  = (_Float16*)(ws + RM0);
  _Float16* dec0   = (_Float16*)(ws + RM0 + 3145728);
  _Float16* dec1   = (_Float16*)(ws + RM0 + 6291456);
  _Float16* comb16 = (_Float16*)(ws + RM1);
  float* hT0 = (float*)(ws + RS0);
  float* cT0 = (float*)(ws + RS0 +  262144);
  float* hT1 = (float*)(ws + RS0 +  524288);
  float* cT1 = (float*)(ws + RS0 +  786432);
  float* hTx = (float*)(ws + RS0 + 1048576);
  float* cTx = (float*)(ws + RS0 + 1310720);

  dim3 blk(256);
  // --- VSN (fused per-feature GRN) -> flat16 (MENC x 1152) ---
  vsn_kernel<<<dim3(MENC_/16, F_), blk, 0, stream>>>(
      x, vsn_fc1_w, vsn_fc1_b, vsn_fc2_w, vsn_fc2_b, vsn_gate_w, vsn_gate_b,
      vsn_skip_w, vsn_skip_b, vsn_ln_s, vsn_ln_b, flat16);
  // --- combined GRN ---
  gemm16<1><<<dim3(MENC_/16,1), blk, 0, stream>>>(flat16, MENC_, FH_, comb_fc1_w, H_, 0, comb_fc1_b, nullptr, nullptr, h1c16, H_);
  gemm16<0><<<dim3(MENC_/16,1), blk, 0, stream>>>(h1c16, MENC_, H_, comb_fc2_w, H_, 0, comb_fc2_b, nullptr, hF, nullptr, H_);
  gemm16<2><<<dim3(MENC_/16,1), blk, 0, stream>>>(h1c16, MENC_, H_, comb_gate_w, H_, 0, comb_gate_b, nullptr, gF, nullptr, H_);
  gemm16<0><<<dim3(MENC_/16,1), blk, 0, stream>>>(flat16, MENC_, FH_, comb_skip_w, H_, 0, comb_skip_b, nullptr, sF, nullptr, H_);
  ln_combine<<<MENC_/8, blk, 0, stream>>>(sF, hF, gF, comb_ln_s, comb_ln_b, MENC_, comb16, nullptr);
  // --- encoder LSTM layer 0 ---
  gemm16<0><<<dim3(MENC_/16,4), blk, 0, stream>>>(comb16, MENC_, H_, enc_Wih, H_, 1, enc_bih, enc_bhh, Gbuf, nullptr, 512);
  lstm_rec<<<B_/16, blk, 0, stream>>>(Gbuf, enc_Whh, nullptr, nullptr, TENC_, seq0, hT0, cT0);
  // --- encoder LSTM layer 1 ---
  gemm16<0><<<dim3(MENC_/16,4), blk, 0, stream>>>(seq0, MENC_, H_, enc_Wih + 512*H_, H_, 1, enc_bih + 512, enc_bhh + 512, Gbuf, nullptr, 512);
  lstm_rec<<<B_/16, blk, 0, stream>>>(Gbuf, enc_Whh + 512*H_, nullptr, nullptr, TENC_, enc1, hT1, cT1);
  // --- decoder ---
  repeat_last<<<(MDEC_*128)/256, blk, 0, stream>>>(enc1, decin);
  gemm16<0><<<dim3(MDEC_/16,4), blk, 0, stream>>>(decin, MDEC_, H_, dec_Wih, H_, 1, dec_bih, dec_bhh, Gbuf, nullptr, 512);
  lstm_rec<<<B_/16, blk, 0, stream>>>(Gbuf, dec_Whh, hT0, cT0, TDEC_, dec0, hTx, cTx);
  gemm16<0><<<dim3(MDEC_/16,4), blk, 0, stream>>>(dec0, MDEC_, H_, dec_Wih + 512*H_, H_, 1, dec_bih + 512, dec_bhh + 512, Gbuf, nullptr, 512);
  lstm_rec<<<B_/16, blk, 0, stream>>>(Gbuf, dec_Whh + 512*H_, hT1, cT1, TDEC_, dec1, hTx, cTx);
  // --- attention projections (Wq/Wk/Wv are rows of qkv_w, transposed-B) ---
  gemm16<0><<<dim3(MDEC_/16,1), blk, 0, stream>>>(dec1, MDEC_, H_, qkv_w,          H_, 1, qkv_b,       nullptr, nullptr, q16, H_);
  gemm16<0><<<dim3(MENC_/16,1), blk, 0, stream>>>(enc1, MENC_, H_, qkv_w + 128*H_, H_, 1, qkv_b + 128, nullptr, nullptr, k16, H_);
  gemm16<0><<<dim3(MENC_/16,1), blk, 0, stream>>>(enc1, MENC_, H_, qkv_w + 256*H_, H_, 1, qkv_b + 256, nullptr, nullptr, v16, H_);
  attn_core<<<MDEC_, dim3(128), 0, stream>>>(q16, k16, v16, ctx16);
  gemm16<0><<<dim3(MDEC_/16,1), blk, 0, stream>>>(ctx16, MDEC_, H_, attn_out_w, H_, 1, attn_out_b, nullptr, attnF, attn16, H_);
  // --- post GRN ---
  gemm16<1><<<dim3(MDEC_/16,1), blk, 0, stream>>>(attn16, MDEC_, H_, post_fc1_w, H_, 0, post_fc1_b, nullptr, nullptr, ph116, H_);
  gemm16<0><<<dim3(MDEC_/16,1), blk, 0, stream>>>(ph116, MDEC_, H_, post_fc2_w, H_, 0, post_fc2_b, nullptr, phF, nullptr, H_);
  gemm16<2><<<dim3(MDEC_/16,1), blk, 0, stream>>>(ph116, MDEC_, H_, post_gate_w, H_, 0, post_gate_b, nullptr, pgF, nullptr, H_);
  ln_combine<<<MDEC_/8, blk, 0, stream>>>(attnF, phF, pgF, post_ln_s, post_ln_b, MDEC_, post16, nullptr);
  // --- final projection ---
  final_out<<<(MDEC_*OUT_ + 255)/256, blk, 0, stream>>>(post16, fc_out_w, fc_out_b, (float*)d_out);
}
